// CombCrossAttention_56710748176705
// MI455X (gfx1250) — matrix operand order repaired and checked by reading, not compiled, not run on hardware
//
#include <hip/hip_runtime.h>

typedef __bf16 bf16;
typedef __attribute__((ext_vector_type(16))) __bf16 bf16x16;
typedef __attribute__((ext_vector_type(8)))  __bf16 bf16x8;
typedef __attribute__((ext_vector_type(4)))  __bf16 bf16x4;
typedef __attribute__((ext_vector_type(8)))  float  f32x8;

// ---------------------------------------------------------------------------
// WMMA helpers (CDNA5 wave32, V_WMMA_F32_16X16X32_BF16)
// A fragment (16x32, MxK): lane m = lane&15, koff = 8*(lane>>4);
//   elems 0..7 = A[m][k0+koff .. +7], elems 8..15 = A[m][k0+16+koff .. +7]
// B fragment (32x16, KxN): lane n = lane&15, koff = 8*(lane>>4);
//   elems 0..7 = B[k0+koff .. +7][n], elems 8..15 = B[k0+16+koff .. +7][n]
// C/D (16x16 f32): reg r, lane l -> row r + 8*(l>>4), col l&15
// ---------------------------------------------------------------------------

static __device__ __forceinline__ bf16x16
load_a_frag(const bf16* __restrict__ base, int row0, long ld, int k0, int lane) {
  const int r    = row0 + (lane & 15);
  const int koff = k0 + ((lane >> 4) << 3);
  const bf16* p  = base + (long)r * ld + koff;
  bf16x8 lo = *(const bf16x8*)(p);
  bf16x8 hi = *(const bf16x8*)(p + 16);
  return __builtin_shufflevector(lo, hi, 0,1,2,3,4,5,6,7,8,9,10,11,12,13,14,15);
}

static __device__ __forceinline__ bf16x16
load_b_frag(const bf16* __restrict__ base, int row0, long ld, int k0, int lane) {
  const int r    = row0 + (lane & 15);
  const int koff = k0 + ((lane >> 4) << 3);
  const bf16* p  = base + (long)r * ld + koff;
  bf16x8 lo = *(const bf16x8*)(p);
  bf16x8 hi = *(const bf16x8*)(p + 16);
  return __builtin_shufflevector(lo, hi, 0,1,2,3,4,5,6,7,8,9,10,11,12,13,14,15);
}

static __device__ __forceinline__ f32x8 wmma_bf16(bf16x16 a, bf16x16 b, f32x8 c) {
  return __builtin_amdgcn_wmma_f32_16x16x32_bf16(false, a, false, b, (short)0, c,
                                                 false, false);
}

// ---------------------------------------------------------------------------
// Elementwise fp32 -> bf16 convert
// ---------------------------------------------------------------------------
__global__ __launch_bounds__(256) void cvt_f32_bf16_kernel(
    const float* __restrict__ in, bf16* __restrict__ out, long n) {
  long i = (long)blockIdx.x * blockDim.x + threadIdx.x;
  const long stride = (long)gridDim.x * blockDim.x;
  for (; i < n; i += stride) out[i] = (bf16)in[i];
}

// ---------------------------------------------------------------------------
// V: [K, KVH, D] fp32 -> Vt: [KVH, D, K] bf16  (transposed so PV B-fragments
// are 16B loads along the key axis)
// ---------------------------------------------------------------------------
__global__ __launch_bounds__(256) void vtrans_kernel(
    const float* __restrict__ v, bf16* __restrict__ vt, int KK, int KVH, int D) {
  const long n = (long)KK * KVH * D;
  long i = (long)blockIdx.x * blockDim.x + threadIdx.x;
  const long stride = (long)gridDim.x * blockDim.x;
  for (; i < n; i += stride) {
    const int  d    = (int)(i % D);
    const long r    = i / D;
    const int  kv   = (int)(r % KVH);
    const int  kpos = (int)(r / KVH);
    vt[((long)kv * D + d) * KK + kpos] = (bf16)v[i];
  }
}

// ---------------------------------------------------------------------------
// Per-head RMSNorm of Q ([T,E] fp32, head h at cols h*128) -> qn [H][T][128]
// bf16, with 1/sqrt(D) attention scale folded in. One wave per (t,h), D=128.
// ---------------------------------------------------------------------------
__global__ __launch_bounds__(256) void rmsnorm_q_kernel(
    const float* __restrict__ Q, const float* __restrict__ w,
    bf16* __restrict__ qn, int T, int H, float scale) {
  const int gw   = blockIdx.x * 8 + (threadIdx.x >> 5);
  const int lane = threadIdx.x & 31;
  const int t = gw / H, h = gw % H;
  if (t >= T) return;
  const float4 x = *(const float4*)(Q + ((long)t * H + h) * 128 + lane * 4);
  float ss = x.x * x.x + x.y * x.y + x.z * x.z + x.w * x.w;
  #pragma unroll
  for (int sh = 1; sh < 32; sh <<= 1) ss += __shfl_xor(ss, sh, 32);
  const float inv = rsqrtf(ss * (1.0f / 128.0f) + 1e-5f) * scale;
  const float4 wv = *(const float4*)(w + lane * 4);
  bf16x4 o = { (bf16)(x.x * inv * wv.x), (bf16)(x.y * inv * wv.y),
               (bf16)(x.z * inv * wv.z), (bf16)(x.w * inv * wv.w) };
  *(bf16x4*)(qn + ((long)h * T + t) * 128 + lane * 4) = o;
}

// ---------------------------------------------------------------------------
// RMSNorm of K ([K,KVH,D] fp32) -> kn [KVH][K][128] bf16. One wave per row.
// ---------------------------------------------------------------------------
__global__ __launch_bounds__(256) void rmsnorm_k_kernel(
    const float* __restrict__ k, const float* __restrict__ w,
    bf16* __restrict__ kn, int KK, int KVH) {
  const int gw   = blockIdx.x * 8 + (threadIdx.x >> 5);
  const int lane = threadIdx.x & 31;
  const int kv = gw / KK, kpos = gw % KK;
  if (kv >= KVH) return;
  const float4 x = *(const float4*)(k + ((long)kpos * KVH + kv) * 128 + lane * 4);
  float ss = x.x * x.x + x.y * x.y + x.z * x.z + x.w * x.w;
  #pragma unroll
  for (int sh = 1; sh < 32; sh <<= 1) ss += __shfl_xor(ss, sh, 32);
  const float inv = rsqrtf(ss * (1.0f / 128.0f) + 1e-5f);
  const float4 wv = *(const float4*)(w + lane * 4);
  bf16x4 o = { (bf16)(x.x * inv * wv.x), (bf16)(x.y * inv * wv.y),
               (bf16)(x.z * inv * wv.z), (bf16)(x.w * inv * wv.w) };
  *(bf16x4*)(kn + ((long)kv * KK + kpos) * 128 + lane * 4) = o;
}

// ---------------------------------------------------------------------------
// C[M,N] (f32) = A[M,K] @ B[N,K]^T, A/B bf16 row-major.
// Block = 256 threads (8 waves) -> 128x128 C tile; wave = 32x64 (2x4 WMMA C).
// Operands stream from global (whole problem is L2-resident: 192MB L2).
// ---------------------------------------------------------------------------
__global__ __launch_bounds__(256) void gemm_bf16_nt_kernel(
    const bf16* __restrict__ A, const bf16* __restrict__ B,
    float* __restrict__ C, int M, int N, int K) {
  const int lane = threadIdx.x & 31;
  const int wave = threadIdx.x >> 5;
  const int m0 = blockIdx.y * 128 + (wave >> 1) * 32;
  const int n0 = blockIdx.x * 128 + (wave & 1) * 64;

  f32x8 acc[2][4];
  #pragma unroll
  for (int i = 0; i < 2; ++i)
    #pragma unroll
    for (int j = 0; j < 4; ++j) acc[i][j] = (f32x8){0,0,0,0,0,0,0,0};

  for (int k0 = 0; k0 < K; k0 += 32) {
    bf16x16 a0 = load_a_frag(A, m0,      K, k0, lane);
    bf16x16 a1 = load_a_frag(A, m0 + 16, K, k0, lane);
    __builtin_prefetch(A + (long)(m0 + (lane & 15)) * K + k0 + 64, 0, 1);
    bf16x16 b0 = load_b_frag(B, n0,      K, k0, lane);
    bf16x16 b1 = load_b_frag(B, n0 + 16, K, k0, lane);
    bf16x16 b2 = load_b_frag(B, n0 + 32, K, k0, lane);
    bf16x16 b3 = load_b_frag(B, n0 + 48, K, k0, lane);
    __builtin_prefetch(B + (long)(n0 + (lane & 15)) * K + k0 + 64, 0, 1);
    acc[0][0] = wmma_bf16(a0, b0, acc[0][0]);
    acc[0][1] = wmma_bf16(a0, b1, acc[0][1]);
    acc[0][2] = wmma_bf16(a0, b2, acc[0][2]);
    acc[0][3] = wmma_bf16(a0, b3, acc[0][3]);
    acc[1][0] = wmma_bf16(a1, b0, acc[1][0]);
    acc[1][1] = wmma_bf16(a1, b1, acc[1][1]);
    acc[1][2] = wmma_bf16(a1, b2, acc[1][2]);
    acc[1][3] = wmma_bf16(a1, b3, acc[1][3]);
  }

  const int half = lane >> 4, nl = lane & 15;
  #pragma unroll
  for (int i = 0; i < 2; ++i)
    #pragma unroll
    for (int j = 0; j < 4; ++j)
      #pragma unroll
      for (int r = 0; r < 8; ++r)
        C[(long)(m0 + 16 * i + r + 8 * half) * N + (n0 + 16 * j + nl)] =
            acc[i][j][r];
}

// ---------------------------------------------------------------------------
// Flash attention (non-causal). One wave per (head, 16-query tile).
// qn [H][T][128], kn [KVH][K][128], vt [KVH][128][K]; output bf16 [T][E].
// Per 32-key step: 8 WMMAs QK^T, online softmax (cross-lane reductions within
// 16-lane halves), P staged C-layout -> A-layout through LDS, 8 WMMAs P*V.
// ---------------------------------------------------------------------------
__global__ __launch_bounds__(256) void attn_kernel(
    const bf16* __restrict__ qn, const bf16* __restrict__ kn,
    const bf16* __restrict__ vt, bf16* __restrict__ ob,
    int T, int KK, int H, int KVH) {
  __shared__ bf16 ptile_s[8][16 * 32];
  const int lane = threadIdx.x & 31;
  const int wave = threadIdx.x >> 5;
  const int gw   = blockIdx.x * 8 + wave;
  const int ntb  = T >> 4;
  const int h    = gw / ntb;
  const int tb   = gw % ntb;
  const int kvh  = h / (H / KVH);       // GQA repeat_interleave mapping
  const int E    = H * 128;
  const bf16* Q  = qn + (long)h   * T  * 128;
  const bf16* Km = kn + (long)kvh * KK * 128;
  const bf16* Vt = vt + (long)kvh * 128 * KK;
  const int t0   = tb * 16;
  const int half = lane >> 4, nl = lane & 15;
  bf16* tile = ptile_s[wave];

  bf16x16 qf[4];                                  // Q tile 16x128 (scale folded)
  #pragma unroll
  for (int dc = 0; dc < 4; ++dc) qf[dc] = load_a_frag(Q, t0, 128, dc * 32, lane);

  f32x8 o[8];                                     // O accumulator 16x128
  float rmax[8], rsum[8], alpha[8];
  #pragma unroll
  for (int j = 0; j < 8; ++j) o[j] = (f32x8){0,0,0,0,0,0,0,0};
  #pragma unroll
  for (int r = 0; r < 8; ++r) { rmax[r] = -1e30f; rsum[r] = 0.0f; }

  for (int key0 = 0; key0 < KK; key0 += 32) {
    // ---- scores S = Q K^T for 32 keys (two 16-wide n tiles) ----
    f32x8 s0 = {0,0,0,0,0,0,0,0}, s1 = {0,0,0,0,0,0,0,0};
    #pragma unroll
    for (int dc = 0; dc < 4; ++dc) {
      s0 = wmma_bf16(qf[dc], load_b_frag(Km, key0,      128, dc * 32, lane), s0);
      s1 = wmma_bf16(qf[dc], load_b_frag(Km, key0 + 16, 128, dc * 32, lane), s1);
    }
    // ---- online softmax: rows live across one 16-lane half ----
    #pragma unroll
    for (int r = 0; r < 8; ++r) {
      float sm = fmaxf(s0[r], s1[r]);
      #pragma unroll
      for (int sh = 1; sh < 16; sh <<= 1) sm = fmaxf(sm, __shfl_xor(sm, sh, 32));
      const float nm = fmaxf(rmax[r], sm);
      alpha[r] = __expf(rmax[r] - nm);
      rmax[r]  = nm;
      const float p0 = __expf(s0[r] - nm);
      const float p1 = __expf(s1[r] - nm);
      float ps = p0 + p1;
      #pragma unroll
      for (int sh = 1; sh < 16; sh <<= 1) ps += __shfl_xor(ps, sh, 32);
      rsum[r] = rsum[r] * alpha[r] + ps;
      s0[r] = p0; s1[r] = p1;                     // reuse regs to hold P
    }
    #pragma unroll
    for (int j = 0; j < 8; ++j)
      #pragma unroll
      for (int r = 0; r < 8; ++r) o[j][r] *= alpha[r];

    // ---- stage P (C-layout) into LDS, reload as A-fragment ----
    #pragma unroll
    for (int r = 0; r < 8; ++r) {
      const int m = r + 8 * half;
      tile[m * 32 + nl]      = (bf16)s0[r];
      tile[m * 32 + 16 + nl] = (bf16)s1[r];
    }
    __syncthreads();
    bf16x16 pf;
    {
      const bf16* pp = tile + (lane & 15) * 32 + ((lane >> 4) << 3);
      bf16x8 lo = *(const bf16x8*)(pp);
      bf16x8 hi = *(const bf16x8*)(pp + 16);
      pf = __builtin_shufflevector(lo, hi,
                                   0,1,2,3,4,5,6,7,8,9,10,11,12,13,14,15);
    }
    __syncthreads();

    // ---- O += P @ V  (contract 32 keys; 8 d-tiles of 16) ----
    #pragma unroll
    for (int j = 0; j < 8; ++j)
      o[j] = wmma_bf16(pf, load_b_frag(Vt, j * 16, KK, key0, lane), o[j]);
  }

  #pragma unroll
  for (int r = 0; r < 8; ++r) rsum[r] = 1.0f / rsum[r];
  #pragma unroll
  for (int j = 0; j < 8; ++j)
    #pragma unroll
    for (int r = 0; r < 8; ++r) {
      const int t = t0 + r + 8 * half;
      const int d = j * 16 + nl;
      ob[(long)t * E + h * 128 + d] = (bf16)(o[j][r] * rsum[r]);
    }
}

// ---------------------------------------------------------------------------
// Host-side orchestration
// ---------------------------------------------------------------------------
extern "C" void kernel_launch(void* const* d_in, const int* in_sizes, int n_in,
                              void* d_out, int out_size, void* d_ws, size_t ws_size,
                              hipStream_t stream) {
  (void)in_sizes; (void)n_in; (void)out_size; (void)ws_size;
  const float* hs = (const float*)d_in[0];   // [T, E]
  const float* kI = (const float*)d_in[1];   // [1, K, KVH, D]
  const float* vI = (const float*)d_in[2];   // [1, K, KVH, D]
  const float* Wq = (const float*)d_in[3];   // [E, E]
  const float* Wo = (const float*)d_in[4];   // [E, E]
  const float* qw = (const float*)d_in[5];   // [D]
  const float* kw = (const float*)d_in[6];   // [D]
  float* out = (float*)d_out;                // [T, E]

  const int T = 2048, KK = 2048, E = 4096, H = 32, KVH = 8;
  const float scale = 0.08838834764831845f;  // 128^-0.5, folded into Q

  char* ws = (char*)d_ws;
  size_t off = 0;
  auto carve = [&](size_t bytes) -> void* {
    void* p = ws + off;
    off += (bytes + 255) & ~(size_t)255;
    return p;
  };
  bf16*  hb  = (bf16*) carve((size_t)T * E * 2);        // hidden bf16
  bf16*  wqb = (bf16*) carve((size_t)E * E * 2);        // Wq bf16
  bf16*  wob = (bf16*) carve((size_t)E * E * 2);        // Wo bf16
  float* Qf  = (float*)carve((size_t)T * E * 4);        // Q = hs @ Wq^T (f32)
  bf16*  qn  = (bf16*) carve((size_t)H * T * 128 * 2);  // normed Q [H][T][D]
  bf16*  kn  = (bf16*) carve((size_t)KVH * KK * 128 * 2); // normed K [KVH][K][D]
  bf16*  vt  = (bf16*) carve((size_t)KVH * 128 * KK * 2); // V^T [KVH][D][K]
  bf16*  ab  = (bf16*) carve((size_t)T * E * 2);        // attn out bf16 [T][E]

  cvt_f32_bf16_kernel<<<1024, 256, 0, stream>>>(hs, hb, (long)T * E);
  cvt_f32_bf16_kernel<<<2048, 256, 0, stream>>>(Wq, wqb, (long)E * E);
  cvt_f32_bf16_kernel<<<2048, 256, 0, stream>>>(Wo, wob, (long)E * E);
  vtrans_kernel<<<1024, 256, 0, stream>>>(vI, vt, KK, KVH, 128);
  rmsnorm_k_kernel<<<(KK * KVH) / 8, 256, 0, stream>>>(kI, kw, kn, KK, KVH);

  gemm_bf16_nt_kernel<<<dim3(E / 128, T / 128), 256, 0, stream>>>(
      hb, wqb, Qf, T, E, E);
  rmsnorm_q_kernel<<<(T * H) / 8, 256, 0, stream>>>(Qf, qw, qn, T, H, scale);

  attn_kernel<<<(H * (T / 16)) / 8, 256, 0, stream>>>(
      qn, kn, vt, ab, T, KK, H, KVH);

  gemm_bf16_nt_kernel<<<dim3(E / 128, T / 128), 256, 0, stream>>>(
      ab, wob, out, T, E, E);
}
